// Architecture_1365799600733
// MI455X (gfx1250) — matrix operand order repaired and checked by reading, not compiled, not run on hardware
//
#include <hip/hip_runtime.h>

typedef _Float16 half_t;
typedef __attribute__((ext_vector_type(16))) _Float16 v16h;
typedef __attribute__((ext_vector_type(8)))  float    v8f;
typedef __attribute__((ext_vector_type(4)))  unsigned int uint4v;
typedef __attribute__((ext_vector_type(4)))  unsigned int u32x4;
typedef __attribute__((ext_vector_type(8)))  int i32x8;
typedef __attribute__((ext_vector_type(4)))  int i32x4;

#define B_  16
#define S_  512
#define D_  1024
#define H_  16
#define L_  4
#define FF_ 4096
#define DK_ 64
#define EPS_ 1e-5f
#define NEG_ -1e32f

#if defined(__gfx1250__) && __has_builtin(__builtin_amdgcn_tensor_load_to_lds)
#define USE_TDM 1
#else
#define USE_TDM 0
#endif

union Frag16 { v16h h; uint4v u[2]; };

// Wave-local LDS fence: DS ops from one wave complete in order; the asm
// memory clobber + wave_barrier stop the compiler reordering around it.
__device__ inline void wave_sync_lds() {
  __builtin_amdgcn_wave_barrier();
  asm volatile("s_wait_dscnt 0" ::: "memory");
  __builtin_amdgcn_wave_barrier();
}

// A-operand (16x32 f16, row-major with leading dim `ld` halfs).
// ISA layout: lanes 0-15 hold M=lane, K {0..7, 16..23}; lanes 16-31 hold
// M=lane-16, K {8..15, 24..31}. Works for both LDS and global pointers.
__device__ inline v16h load_a_frag(const half_t* t, int ld) {
  int lane = threadIdx.x & 31;
  int m = lane & 15;
  int koff = (lane >> 4) << 3;
  Frag16 f;
  f.u[0] = *(const uint4v*)(t + (size_t)m * ld + koff);
  f.u[1] = *(const uint4v*)(t + (size_t)m * ld + koff + 16);
  return f.h;
}

// B-operand (32x16 f16) from a transposed view tileT[n][k] with leading dim
// `ld`: lanes 0-15 hold N=lane, K {0..7, 16..23}; lanes 16-31 hold K {8..15, 24..31}.
__device__ inline v16h load_b_frag(const half_t* t, int ld) {
  int lane = threadIdx.x & 31;
  int n = lane & 15;
  int koff = (lane >> 4) << 3;
  Frag16 f;
  f.u[0] = *(const uint4v*)(t + (size_t)n * ld + koff);
  f.u[1] = *(const uint4v*)(t + (size_t)n * ld + koff + 16);
  return f.h;
}

#if USE_TDM
// Issue a TDM load of a 16-row x 32-half (2-byte elements) tile whose rows
// stride K halfs in global memory, into contiguous LDS (16x32).
// D# layout per CDNA5 ISA ch.8: group0 {count=1, lds_addr, global_addr,
// type=2}, group1 {data_size=1(2B), tensor_dim0=K, tensor_dim1=16,
// tile_dim0=32, tile_dim1=16, tensor_dim0_stride=K}.
__device__ inline void tdm_load_a_tile(const half_t* gsrc, unsigned ldsAddr, int K) {
  unsigned long long ga = (unsigned long long)(size_t)gsrc;
  u32x4 g0;
  g0.x = 1u;                                                  // count=1 (valid)
  g0.y = ldsAddr;                                             // LDS byte addr
  g0.z = (unsigned)ga;                                        // global addr lo
  g0.w = ((unsigned)(ga >> 32) & 0x1FFFFFFu) | (2u << 30);    // addr hi | type=2
  i32x8 g1;
  g1[0] = (int)(1u << 16);                          // data_size = 2 bytes
  g1[1] = (int)(((unsigned)K & 0xFFFFu) << 16);     // tensor_dim0 lo16
  g1[2] = (int)((((unsigned)K >> 16) & 0xFFFFu) | (16u << 16)); // d0 hi | dim1 lo
  g1[3] = (int)(32u << 16);                         // tensor_dim1 hi | tile_dim0=32
  g1[4] = 16;                                       // tile_dim1=16, tile_dim2=0
  g1[5] = K;                                        // tensor_dim0_stride lo32
  g1[6] = 0;                                        // stride hi | dim1_stride lo
  g1[7] = 0;                                        // dim1_stride hi
  i32x4 z4 = {0, 0, 0, 0};
#if defined(__clang_major__) && __clang_major__ >= 23
  i32x8 z8 = {0, 0, 0, 0, 0, 0, 0, 0};
  __builtin_amdgcn_tensor_load_to_lds(g0, g1, z4, z4, z8, 0);
#else
  __builtin_amdgcn_tensor_load_to_lds(g0, g1, z4, z4, 0);
#endif
}
#endif

// ---------------------------------------------------------------------------
// C[M,N] = A[M,K](f16) @ Wt[N,K](f16, pre-transposed) + bias, optional relu.
// One wave per 16x64 output strip: A-frag reused across 4 WMMAs per K-step.
// A tile staged via TDM (tensor_load_to_lds) when available, else direct
// global fragment loads. B operands load directly from global (L2-resident).
// ---------------------------------------------------------------------------
__global__ __launch_bounds__(256)
void gemm_wmma_kernel(const half_t* __restrict__ A, const half_t* __restrict__ Wt,
                      const float* __restrict__ bias, float* __restrict__ Cf,
                      half_t* __restrict__ Ch, int M, int N, int K, int relu,
                      int totalTiles)
{
#if USE_TDM
  __shared__ __align__(16) half_t sA[8][16 * 32];
#endif
  int wave = threadIdx.x >> 5, lane = threadIdx.x & 31;
  int tid = blockIdx.x * 8 + wave;
  if (tid >= totalTiles) return;
  int ntn = N >> 6;                 // N / 64
  int mt = tid / ntn, nt = tid % ntn;
  v8f acc[4] = {};
  const half_t* Abase = A + (size_t)mt * 16 * K;
  const half_t* Bbase[4];
  #pragma unroll
  for (int c = 0; c < 4; ++c)
    Bbase[c] = Wt + (size_t)(nt * 64 + c * 16) * K;
#if USE_TDM
  half_t* sAw = sA[wave];
  unsigned ldsA = (unsigned)(size_t)(void*)sAw;
#endif
  for (int kk = 0; kk < K; kk += 32) {
#if USE_TDM
    wave_sync_lds();                      // prior a-frag DS reads done
    tdm_load_a_tile(Abase + kk, ldsA, K); // async: TENSORcnt++
#endif
    v16h bfr[4];
    #pragma unroll
    for (int c = 0; c < 4; ++c)
      bfr[c] = load_b_frag(Bbase[c] + kk, K);
    if (kk + 32 < K) {
      __builtin_prefetch(Abase + (size_t)(lane & 15) * K + kk + 32, 0, 1);
      __builtin_prefetch(Bbase[0] + (size_t)(lane & 15) * K + kk + 32, 0, 1);
    }
#if USE_TDM
    __builtin_amdgcn_s_wait_tensorcnt(0);
    __builtin_amdgcn_wave_barrier();
    v16h a = load_a_frag(sAw, 32);
#else
    v16h a = load_a_frag(Abase + kk, K);
#endif
    #pragma unroll
    for (int c = 0; c < 4; ++c)
      acc[c] = __builtin_amdgcn_wmma_f32_16x16x32_f16(false, a, false, bfr[c],
                                                      (short)0, acc[c],
                                                      false, false);
  }
  int n = lane & 15, hh = lane >> 4;
  #pragma unroll
  for (int c = 0; c < 4; ++c) {
    float bn = bias ? bias[nt * 64 + c * 16 + n] : 0.0f;
    #pragma unroll
    for (int i = 0; i < 8; ++i) {
      float v = acc[c][i] + bn;
      if (relu) v = fmaxf(v, 0.0f);
      size_t idx = (size_t)(mt * 16 + i + 8 * hh) * N + nt * 64 + c * 16 + n;
      if (Cf) Cf[idx] = v;
      if (Ch) Ch[idx] = (half_t)v;
    }
  }
}

// ---------------------------------------------------------------------------
// Flash attention, one wave per (b, h, 16-row query tile). q == k (kq_same).
// Q/K fragments load directly from global (row-major layout matches operand
// layouts with ld=D). V is transposed through LDS; P does the C->A layout
// round-trip through LDS. Strictly-causal mask (j < i), forget_rate scaling,
// query row 0 zeroed.
// ---------------------------------------------------------------------------
__global__ __launch_bounds__(256)
void attn_wmma_kernel(const half_t* __restrict__ kh, const half_t* __restrict__ vh,
                      const float* __restrict__ fr, half_t* __restrict__ outh)
{
  __shared__ __align__(16) half_t sVt[8][64 * 32];
  __shared__ __align__(16) half_t sP[8][16 * 32];
  int wave = threadIdx.x >> 5, lane = threadIdx.x & 31;
  int tid = blockIdx.x * 8 + wave;
  const int mtS = S_ / 16;
  int bh = tid / mtS, mt = tid % mtS;
  int b = bh / H_, h = bh % H_;
  half_t* Vt = sVt[wave];
  half_t* P  = sP[wave];
  const half_t* Qbase = kh + (size_t)(b * S_ + mt * 16) * D_ + h * DK_;
  // Hoist Q fragments (reused by every j-block).
  v16h qf[2];
  #pragma unroll
  for (int t = 0; t < 2; ++t) qf[t] = load_a_frag(Qbase + 32 * t, D_);

  int n = lane & 15, hh = lane >> 4;
  float mi[8], li[8], frv[8];
  #pragma unroll
  for (int i = 0; i < 8; ++i) {
    mi[i] = -1e30f;
    li[i] = 0.0f;
    frv[i] = fr[b * S_ + mt * 16 + i + 8 * hh];
  }
  v8f o[4] = {};
  const float scale = 0.125f;  // 1/sqrt(64)
  int jmax = mt * 16 + 14;     // largest allowed key index for this tile
  for (int j0 = 0; j0 <= jmax; j0 += 32) {
    // Stage V rows transposed: Vt[d][j].
    {
      const half_t* gv = vh + (size_t)(b * S_ + j0 + lane) * D_ + h * DK_;
      uint4v tv[8];
      #pragma unroll
      for (int q = 0; q < 8; ++q) tv[q] = *(const uint4v*)(gv + q * 8);
      const half_t* tvh = (const half_t*)tv;
      #pragma unroll
      for (int d = 0; d < 64; ++d) Vt[d * 32 + lane] = tvh[d];
    }
    // Scores: S[16x32] = Q (16x64) @ K^T; row-major K IS the B operand
    // transposed view (B[n=j][k=d] = K[j][d]), loaded straight from global.
    const half_t* Kbase = kh + (size_t)(b * S_ + j0) * D_ + h * DK_;
    v8f sc2[2] = {};
    #pragma unroll
    for (int tj = 0; tj < 2; ++tj) {
      #pragma unroll
      for (int t = 0; t < 2; ++t) {
        v16h bb = load_b_frag(Kbase + (size_t)tj * 16 * D_ + 32 * t, D_);
        sc2[tj] = __builtin_amdgcn_wmma_f32_16x16x32_f16(false, qf[t], false, bb,
                                                         (short)0, sc2[tj],
                                                         false, false);
      }
    }
    // Online softmax update over the 16x32 score block.
    #pragma unroll
    for (int i = 0; i < 8; ++i) {
      int ig = mt * 16 + i + 8 * hh;
      float s0 = sc2[0][i] * scale * frv[i];
      float s1 = sc2[1][i] * scale * frv[i];
      if (j0 + n >= ig)      s0 = NEG_;
      if (j0 + 16 + n >= ig) s1 = NEG_;
      float rm = fmaxf(s0, s1);
      rm = fmaxf(rm, __shfl_xor(rm, 1, 16));
      rm = fmaxf(rm, __shfl_xor(rm, 2, 16));
      rm = fmaxf(rm, __shfl_xor(rm, 4, 16));
      rm = fmaxf(rm, __shfl_xor(rm, 8, 16));
      float mn = fmaxf(mi[i], rm);
      float corr = __expf(mi[i] - mn);
      float p0 = __expf(s0 - mn);
      float p1 = __expf(s1 - mn);
      float rs = p0 + p1;
      rs += __shfl_xor(rs, 1, 16);
      rs += __shfl_xor(rs, 2, 16);
      rs += __shfl_xor(rs, 4, 16);
      rs += __shfl_xor(rs, 8, 16);
      li[i] = li[i] * corr + rs;
      mi[i] = mn;
      #pragma unroll
      for (int c = 0; c < 4; ++c) o[c][i] *= corr;
      P[(i + 8 * hh) * 32 + n]      = (half_t)p0;
      P[(i + 8 * hh) * 32 + 16 + n] = (half_t)p1;
    }
    wave_sync_lds();
    // O += P (16x32) @ V (32x64): 4 WMMAs over d-chunks of 16.
    #pragma unroll
    for (int c = 0; c < 4; ++c) {
      v16h a  = load_a_frag(P, 32);
      v16h bb = load_b_frag(Vt + c * 16 * 32, 32);
      o[c] = __builtin_amdgcn_wmma_f32_16x16x32_f16(false, a, false, bb,
                                                    (short)0, o[c], false, false);
    }
    wave_sync_lds();
  }
  #pragma unroll
  for (int i = 0; i < 8; ++i) {
    float inv = 1.0f / fmaxf(li[i], 1e-30f);
    int srow = mt * 16 + i + 8 * hh;
    #pragma unroll
    for (int c = 0; c < 4; ++c) {
      float v = o[c][i] * inv;
      if (srow == 0) v = 0.0f;  // zero_pad of first query row
      outh[(size_t)(b * S_ + srow) * D_ + h * DK_ + c * 16 + n] = (half_t)v;
    }
  }
}

// ---------------------------------------------------------------------------
// x = LayerNorm(x + add) * g + b ; also writes f16 copy. One block per row.
// ---------------------------------------------------------------------------
__global__ __launch_bounds__(256)
void ln_residual_kernel(float* __restrict__ x, const float* __restrict__ add,
                        const float* __restrict__ g, const float* __restrict__ bb,
                        half_t* __restrict__ xh)
{
  __shared__ float red[256];
  int row = blockIdx.x, t = threadIdx.x;
  size_t base = (size_t)row * D_;
  float v[4];
  float s = 0.0f;
  #pragma unroll
  for (int k = 0; k < 4; ++k) {
    int d = t + k * 256;
    v[k] = x[base + d] + add[base + d];
    s += v[k];
  }
  red[t] = s;
  __syncthreads();
  for (int off = 128; off > 0; off >>= 1) {
    if (t < off) red[t] += red[t + off];
    __syncthreads();
  }
  float mean = red[0] / D_;
  __syncthreads();
  s = 0.0f;
  #pragma unroll
  for (int k = 0; k < 4; ++k) { float d = v[k] - mean; s += d * d; }
  red[t] = s;
  __syncthreads();
  for (int off = 128; off > 0; off >>= 1) {
    if (t < off) red[t] += red[t + off];
    __syncthreads();
  }
  float rstd = rsqrtf(red[0] / D_ + EPS_);
  #pragma unroll
  for (int k = 0; k < 4; ++k) {
    int d = t + k * 256;
    float y = (v[k] - mean) * rstd * g[d] + bb[d];
    x[base + d] = y;
    xh[base + d] = (half_t)y;
  }
}

__global__ void add_pe_kernel(const float* __restrict__ q, const float* __restrict__ qa,
                              const float* __restrict__ pe, float* __restrict__ xf,
                              half_t* __restrict__ xh, half_t* __restrict__ yh, int n)
{
  int i = blockIdx.x * blockDim.x + threadIdx.x;
  if (i >= n) return;
  int d = i % D_;
  int s = (i / D_) % S_;
  float p = pe[s * D_ + d];
  float xv = q[i] + p;
  float yv = qa[i] + p;
  xf[i] = xv;
  xh[i] = (half_t)xv;
  yh[i] = (half_t)yv;
}

// src [K][N] f32 -> dst [N][K] f16 (tiled transpose through LDS, 32x32 tiles,
// padded to kill bank conflicts). Grid: (N/32, K/32), 256 threads (32x8).
__global__ __launch_bounds__(256)
void transpose_cvt_kernel(const float* __restrict__ src, half_t* __restrict__ dst,
                          int K, int N)
{
  __shared__ float tile[32][33];
  int n0 = blockIdx.x * 32, k0 = blockIdx.y * 32;
  int tx = threadIdx.x & 31, ty = threadIdx.x >> 5;
  #pragma unroll
  for (int r = 0; r < 4; ++r)
    tile[ty + r * 8][tx] = src[(size_t)(k0 + ty + r * 8) * N + n0 + tx];
  __syncthreads();
  #pragma unroll
  for (int r = 0; r < 4; ++r)
    dst[(size_t)(n0 + ty + r * 8) * K + k0 + tx] = (half_t)tile[tx][ty + r * 8];
}

extern "C" void kernel_launch(void* const* d_in, const int* in_sizes, int n_in,
                              void* d_out, int out_size, void* d_ws, size_t ws_size,
                              hipStream_t stream)
{
  (void)in_sizes; (void)n_in; (void)out_size; (void)ws_size;
  const float* q_embed = (const float*)d_in[0];
  const float* qa_embed = (const float*)d_in[1];
  const float* forget = (const float*)d_in[2];   // [B,1,S,1] -> B*S
  const float* pe = (const float*)d_in[3];       // [1,S,D]
  const float* Wk = (const float*)d_in[4];
  const float* bk = (const float*)d_in[5];
  const float* Wv = (const float*)d_in[6];
  const float* bv = (const float*)d_in[7];
  const float* Wo = (const float*)d_in[8];
  const float* bo = (const float*)d_in[9];
  const float* W1 = (const float*)d_in[10];
  const float* b1 = (const float*)d_in[11];
  const float* W2 = (const float*)d_in[12];
  const float* b2 = (const float*)d_in[13];
  const float* ln1_g = (const float*)d_in[14];
  const float* ln1_b = (const float*)d_in[15];
  const float* ln2_g = (const float*)d_in[16];
  const float* ln2_b = (const float*)d_in[17];

  const size_t BSD = (size_t)B_ * S_ * D_;
  const size_t BSF = (size_t)B_ * S_ * FF_;
  char* ws = (char*)d_ws;
  size_t off = 0;
  auto take = [&](size_t bytes) -> char* {
    char* p = ws + off;
    off = (off + bytes + 255) & ~(size_t)255;
    return p;
  };
  float*  xf   = (float*)take(BSD * 4);
  float*  tmpf = (float*)take(BSD * 4);
  half_t* xh   = (half_t*)take(BSD * 2);
  half_t* yh   = (half_t*)take(BSD * 2);
  half_t* kh   = (half_t*)take(BSD * 2);
  half_t* vh   = (half_t*)take(BSD * 2);
  half_t* aoh  = (half_t*)take(BSD * 2);
  half_t* ffh  = (half_t*)take(BSF * 2);
  half_t* wkh  = (half_t*)take((size_t)D_ * D_ * 2);   // transposed [N][K]
  half_t* wvh  = (half_t*)take((size_t)D_ * D_ * 2);
  half_t* woh  = (half_t*)take((size_t)D_ * D_ * 2);
  half_t* w1h  = (half_t*)take((size_t)D_ * FF_ * 2);  // [FF][D]
  half_t* w2h  = (half_t*)take((size_t)FF_ * D_ * 2);  // [D][FF]

  const int M = B_ * S_;  // 8192
  int n = (int)BSD;
  add_pe_kernel<<<(n + 255) / 256, 256, 0, stream>>>(q_embed, qa_embed, pe, xf, xh, yh, n);

  auto gemm = [&](const half_t* A, const half_t* Wt, const float* bias,
                  float* Cf, half_t* Ch, int N, int K, int relu) {
    int tiles = (M / 16) * (N / 64);
    gemm_wmma_kernel<<<(tiles + 7) / 8, 256, 0, stream>>>(A, Wt, bias, Cf, Ch, M, N, K, relu, tiles);
  };
  auto tcvt = [&](const float* s, half_t* d, int K, int N) {
    dim3 grid(N / 32, K / 32);
    transpose_cvt_kernel<<<grid, 256, 0, stream>>>(s, d, K, N);
  };

  for (int l = 0; l < L_; ++l) {
    tcvt(Wk + (size_t)l * D_ * D_, wkh, D_, D_);
    tcvt(Wv + (size_t)l * D_ * D_, wvh, D_, D_);
    tcvt(Wo + (size_t)l * D_ * D_, woh, D_, D_);
    tcvt(W1 + (size_t)l * D_ * FF_, w1h, D_, FF_);
    tcvt(W2 + (size_t)l * FF_ * D_, w2h, FF_, D_);

    // k = x@Wk + bk (q == k), v = y@Wv + bv
    gemm(xh, wkh, bk + l * D_, nullptr, kh, D_, D_, 0);
    gemm(yh, wvh, bv + l * D_, nullptr, vh, D_, D_, 0);

    // flash attention -> aoh (f16)
    attn_wmma_kernel<<<(B_ * H_ * (S_ / 16)) / 8, 256, 0, stream>>>(kh, vh, forget, aoh);

    // out projection + LN1
    gemm(aoh, woh, bo + l * D_, tmpf, nullptr, D_, D_, 0);
    ln_residual_kernel<<<M, 256, 0, stream>>>(xf, tmpf, ln1_g + l * D_, ln1_b + l * D_, xh);

    // FFN + LN2
    gemm(xh, w1h, b1 + l * FF_, nullptr, ffh, FF_, D_, 1);
    gemm(ffh, w2h, b2 + l * D_, tmpf, nullptr, D_, FF_, 0);
    ln_residual_kernel<<<M, 256, 0, stream>>>(xf, tmpf, ln2_g + l * D_, ln2_b + l * D_, xh);
  }

  hipMemcpyAsync(d_out, xf, BSD * 4, hipMemcpyDeviceToDevice, stream);
}
